// GRUAttnClassifier_33578054320558
// MI455X (gfx1250) — matrix operand order repaired and checked by reading, not compiled, not run on hardware
//
#include <hip/hip_runtime.h>
#include <hip/hip_bf16.h>

// ---------------- problem constants ----------------
#define BATCH 256
#define CIN   64
#define TT    2048
#define HH    64
#define HEADN 32
#define NCLS  8
#define NBT   16          // batch tiles of 16

typedef __attribute__((ext_vector_type(16))) __bf16 v16bf;
typedef __attribute__((ext_vector_type(8)))  float  v8f;

union BF {                // one WMMA 16-bit fragment image (32 bytes / lane)
  v16bf v;
  unsigned short u[16];
  uint4 q[2];
};

__device__ __forceinline__ unsigned short f2bf(float f) {
  unsigned int x = __float_as_uint(f);
  unsigned int r = x + 0x7FFFu + ((x >> 16) & 1u);   // round-to-nearest-even
  return (unsigned short)(r >> 16);
}

__device__ __forceinline__ v8f splat8(float x) {
  v8f v = {x, x, x, x, x, x, x, x};
  return v;
}

__device__ __forceinline__ v8f wmma_bf16(v16bf a, v16bf b, v8f c) {
  return __builtin_amdgcn_wmma_f32_16x16x32_bf16(
      /*neg_a=*/false, a, /*neg_b=*/false, b,
      /*c_mod=*/(short)0, c, /*reuse_a=*/false, /*reuse_b=*/false);
}

__device__ __forceinline__ float sigmoidf_(float x) {
  return 1.0f / (1.0f + __expf(-x));
}

// ==========================================================================
// Recurrent GRU layer. One WG per 16-row batch tile, 4 waves; wave w owns
// hidden columns [16w,16w+16) of all 3 gates. Weights live in VGPR B-frags.
// h is double-buffered in LDS as bf16 in WMMA A-fragment-friendly layout.
// Hidden sequence is written to global in A-fragment layout:
//   hout[((t*16+btile)*2 + frag)*64 + lane*2 + {0,1}]  (uint4 units)
// LAYER==0 : input from x (B,C,T) f32, gathered per step.
// LAYER==1 : input from previous layer's fragment-layout sequence.
// ==========================================================================
template <int LAYER>
__global__ __launch_bounds__(128) void gru_rec_kernel(
    const float* __restrict__ x,        // LAYER==0 input
    const uint4* __restrict__ hin,      // LAYER==1 input (frag layout)
    const float* __restrict__ w_ih,     // (192, 64)
    const float* __restrict__ w_hh,     // (192, 64)
    const float* __restrict__ b_ih,     // (192,)
    const float* __restrict__ b_hh,     // (192,)
    uint4* __restrict__ hout)           // frag layout output
{
  const int btile = blockIdx.x;             // 0..15
  const int wave  = threadIdx.x >> 5;       // 0..3
  const int lane  = threadIdx.x & 31;
  const int m     = lane & 15;              // batch row within tile
  const int half8 = (lane >> 4) << 3;       // 0 / 8 : K-half (A) and M-half (C)
  const int b0    = btile * 16;

  __shared__ unsigned short hbuf[2][16][64];  // h tiles, bf16, double buffered

  // h_{-1} = 0
  for (int idx = threadIdx.x; idx < 16 * 64; idx += 128)
    ((unsigned short*)hbuf[0])[idx] = 0;

  // ---- build persistent B fragments: W^T is (K=64) x (N=192) ----
  // B frag layout: lane%16 = local N, component i = K in table order (as A).
  BF Bih[3][2], Bhh[3][2];
  float bi[3], bh[3];
#pragma unroll
  for (int g = 0; g < 3; ++g) {
    const int nbase = g * 64 + wave * 16;
#pragma unroll
    for (int f = 0; f < 2; ++f) {
#pragma unroll
      for (int i = 0; i < 16; ++i) {
        const int k = f * 32 + ((i >> 3) << 4) + half8 + (i & 7);
        Bih[g][f].u[i] = f2bf(w_ih[(nbase + m) * 64 + k]);
        Bhh[g][f].u[i] = f2bf(w_hh[(nbase + m) * 64 + k]);
      }
    }
    bi[g] = b_ih[nbase + m];
    bh[g] = b_hh[nbase + m];
  }

  v8f hp = {};      // h_{t-1} for this wave's columns (C layout)
  int p = 0;
  __syncthreads();

  for (int t = 0; t < TT; ++t) {
    // ---- A_x fragments (independent of h: issue before the barrier) ----
    BF Ax[2];
    if (LAYER == 0) {
#pragma unroll
      for (int f = 0; f < 2; ++f)
#pragma unroll
        for (int i = 0; i < 16; ++i) {
          const int k = f * 32 + ((i >> 3) << 4) + half8 + (i & 7);
          Ax[f].u[i] =
              f2bf(x[(size_t)(b0 + m) * (CIN * TT) + (size_t)k * TT + t]);
        }
    } else {
      const uint4* s = hin + ((size_t)(t * NBT + btile) * 2) * 64 + lane * 2;
      Ax[0].q[0] = s[0];
      Ax[0].q[1] = s[1];
      Ax[1].q[0] = s[64];
      Ax[1].q[1] = s[65];
    }

    __syncthreads();          // hbuf[p] now holds h_{t-1}

    // ---- A_h fragments from LDS (2x b128 each) ----
    BF Ah[2];
#pragma unroll
    for (int f = 0; f < 2; ++f) {
      Ah[f].q[0] = *(const uint4*)&hbuf[p][m][f * 32 + half8];
      Ah[f].q[1] = *(const uint4*)&hbuf[p][m][f * 32 + 16 + half8];
    }

    // spill h_{t-1} to global sequence (register image == fragment image)
    if (wave < 2 && t > 0) {
      uint4* d =
          hout + (((size_t)((t - 1) * NBT + btile) * 2 + wave) * 64) + lane * 2;
      d[0] = Ah[wave].q[0];
      d[1] = Ah[wave].q[1];
    }

    // ---- 12 WMMAs: xg = x W_ih^T + b_ih ; hg = h W_hh^T + b_hh ----
    v8f xa[3], ha[3];
#pragma unroll
    for (int g = 0; g < 3; ++g) {
      xa[g] = splat8(bi[g]);
      ha[g] = splat8(bh[g]);
      xa[g] = wmma_bf16(Ax[0].v, Bih[g][0].v, xa[g]);
      xa[g] = wmma_bf16(Ax[1].v, Bih[g][1].v, xa[g]);
      ha[g] = wmma_bf16(Ah[0].v, Bhh[g][0].v, ha[g]);
      ha[g] = wmma_bf16(Ah[1].v, Bhh[g][1].v, ha[g]);
    }

    // ---- gate math (wave-local, C layout) ----
    v8f hn;
#pragma unroll
    for (int r = 0; r < 8; ++r) {
      const float rg = sigmoidf_(xa[0][r] + ha[0][r]);
      const float zg = sigmoidf_(xa[1][r] + ha[1][r]);
      const float ng = tanhf(xa[2][r] + rg * ha[2][r]);
      hn[r] = ng + zg * (hp[r] - ng);
    }
    hp = hn;

    // write h_t (bf16) into the other LDS buffer
    const int col = wave * 16 + m;
#pragma unroll
    for (int r = 0; r < 8; ++r)
      hbuf[p ^ 1][r + half8][col] = f2bf(hn[r]);
    p ^= 1;
  }

  __syncthreads();
  if (wave < 2) {   // final h_{T-1}
    BF Af;
    Af.q[0] = *(const uint4*)&hbuf[p][m][wave * 32 + half8];
    Af.q[1] = *(const uint4*)&hbuf[p][m][wave * 32 + 16 + half8];
    uint4* d =
        hout + (((size_t)((TT - 1) * NBT + btile) * 2 + wave) * 64) + lane * 2;
    d[0] = Af.q[0];
    d[1] = Af.q[1];
  }
}

// ==========================================================================
// Attention logits: wlog[b][t] = tanh(h W1^T + b1) . w2 + b2, via WMMA.
// One wave per (btile, 16 t's). Weight fragments loaded once per block.
// ==========================================================================
__global__ __launch_bounds__(32) void attn_logits_kernel(
    const uint4* __restrict__ h1,       // frag layout
    const float* __restrict__ w1,       // (64,64)
    const float* __restrict__ b1v,      // (64,)
    const float* __restrict__ w2,       // (64,)  (attn_w2 row)
    const float* __restrict__ b2,       // (1,)
    float* __restrict__ wlog)           // (B,T)
{
  const int btile = blockIdx.x;     // 16
  const int tch   = blockIdx.y;     // 128
  const int lane  = threadIdx.x;
  const int m     = lane & 15;

  BF Bw[4][2];
  float w2v[4], b1s[4];
#pragma unroll
  for (int nt = 0; nt < 4; ++nt) {
#pragma unroll
    for (int f = 0; f < 2; ++f) {
#pragma unroll
      for (int i = 0; i < 16; ++i) {
        const int k = f * 32 + ((i >> 3) << 4) + ((lane >> 4) << 3) + (i & 7);
        Bw[nt][f].u[i] = f2bf(w1[(nt * 16 + m) * 64 + k]);
      }
    }
    w2v[nt] = w2[nt * 16 + m];
    b1s[nt] = b1v[nt * 16 + m];
  }
  const float bb2 = b2[0];

  for (int tt = 0; tt < 16; ++tt) {
    const int t = tch * 16 + tt;
    BF A0, A1;
    const uint4* s = h1 + ((size_t)(t * NBT + btile) * 2) * 64 + lane * 2;
    A0.q[0] = s[0];  A0.q[1] = s[1];
    A1.q[0] = s[64]; A1.q[1] = s[65];

    v8f sacc = {};
#pragma unroll
    for (int nt = 0; nt < 4; ++nt) {
      v8f c = splat8(b1s[nt]);
      c = wmma_bf16(A0.v, Bw[nt][0].v, c);
      c = wmma_bf16(A1.v, Bw[nt][1].v, c);
#pragma unroll
      for (int r = 0; r < 8; ++r) sacc[r] += tanhf(c[r]) * w2v[nt];
    }
    // reduce over the 16 N-lanes of each half
#pragma unroll
    for (int mask = 1; mask < 16; mask <<= 1)
#pragma unroll
      for (int r = 0; r < 8; ++r) sacc[r] += __shfl_xor(sacc[r], mask, 32);

    if ((lane & 15) == 0) {
      const int mb = (lane >> 4) << 3;
#pragma unroll
      for (int r = 0; r < 8; ++r)
        wlog[(size_t)(btile * 16 + mb + r) * TT + t] = sacc[r] + bb2;
    }
  }
}

// ==========================================================================
// Row softmax over T (one WG of 256 per batch row).
// ==========================================================================
__global__ __launch_bounds__(256) void softmax_kernel(
    const float* __restrict__ wlog, float* __restrict__ wsm)
{
  const int b = blockIdx.x;
  __shared__ float red[256];
  const float* row = wlog + (size_t)b * TT;

  float mx = -1e30f;
  for (int t = threadIdx.x; t < TT; t += 256) mx = fmaxf(mx, row[t]);
  red[threadIdx.x] = mx;
  __syncthreads();
  for (int s = 128; s > 0; s >>= 1) {
    if (threadIdx.x < s)
      red[threadIdx.x] = fmaxf(red[threadIdx.x], red[threadIdx.x + s]);
    __syncthreads();
  }
  mx = red[0];
  __syncthreads();

  float sum = 0.f;
  for (int t = threadIdx.x; t < TT; t += 256) sum += __expf(row[t] - mx);
  red[threadIdx.x] = sum;
  __syncthreads();
  for (int s = 128; s > 0; s >>= 1) {
    if (threadIdx.x < s) red[threadIdx.x] += red[threadIdx.x + s];
    __syncthreads();
  }
  const float inv = 1.f / red[0];
  float* orow = wsm + (size_t)b * TT;
  for (int t = threadIdx.x; t < TT; t += 256)
    orow[t] = __expf(row[t] - mx) * inv;
}

// ==========================================================================
// Weighted pooling over T + LayerNorm + MLP head. One WG (64 thr) per row.
// Reads h1 sequence directly from fragment layout (bf16).
// ==========================================================================
__global__ __launch_bounds__(64) void pool_head_kernel(
    const uint4* __restrict__ h1seq, const float* __restrict__ wsm,
    const float* __restrict__ lng, const float* __restrict__ lnb,
    const float* __restrict__ hw1, const float* __restrict__ hb1,
    const float* __restrict__ hw2, const float* __restrict__ hb2,
    float* __restrict__ out)
{
  const int b   = blockIdx.x;
  const int col = threadIdx.x;           // hidden index 0..63
  const int btile = b >> 4, m = b & 15;

  // invert the A-fragment layout for element (m, col)
  const int f      = col >> 5;
  const int klocal = col & 31;
  const int quad   = (klocal >> 4) & 1;
  const int k8     = klocal & 15;
  const int lane   = m + ((k8 >= 8) ? 16 : 0);
  const int i      = quad * 8 + (k8 & 7);

  const unsigned short* base = (const unsigned short*)h1seq;
  const size_t per_t = (size_t)NBT * 2 * 64 * 8;   // shorts per timestep
  const size_t off0  = (((size_t)btile * 2 + f) * 64 + lane * 2) * 8 + i;

  const float* wr = wsm + (size_t)b * TT;
  float acc = 0.f;
  for (int t = 0; t < TT; ++t) {
    if ((t & 63) == 0 && t + 64 < TT)
      __builtin_prefetch(base + (size_t)(t + 64) * per_t + off0, 0, 1);
    const unsigned short uv = base[(size_t)t * per_t + off0];
    const float hv = __uint_as_float(((unsigned int)uv) << 16);
    acc += wr[t] * hv;
  }

  __shared__ float pv[64], yv[64], gv[32];
  pv[col] = acc;
  __syncthreads();

  float mu = 0.f;
  for (int j = 0; j < 64; ++j) mu += pv[j];
  mu *= (1.f / 64.f);
  float var = 0.f;
  for (int j = 0; j < 64; ++j) { const float d = pv[j] - mu; var += d * d; }
  var *= (1.f / 64.f);
  const float y = (acc - mu) * rsqrtf(var + 1e-5f) * lng[col] + lnb[col];
  yv[col] = y;
  __syncthreads();

  if (col < HEADN) {
    float a = hb1[col];
    for (int j = 0; j < 64; ++j) a += hw1[col * 64 + j] * yv[j];
    gv[col] = 0.5f * a * (1.f + erff(a * 0.70710678118654752f));  // exact GELU
  }
  __syncthreads();

  if (col < NCLS) {
    float o = hb2[col];
    for (int j = 0; j < HEADN; ++j) o += hw2[col * HEADN + j] * gv[j];
    out[(size_t)b * NCLS + col] = o;
  }
}

// ==========================================================================
extern "C" void kernel_launch(void* const* d_in, const int* in_sizes, int n_in,
                              void* d_out, int out_size, void* d_ws,
                              size_t ws_size, hipStream_t stream) {
  (void)in_sizes; (void)n_in; (void)out_size; (void)ws_size;
  const float* x       = (const float*)d_in[0];
  const float* w_ih0   = (const float*)d_in[1];
  const float* w_hh0   = (const float*)d_in[2];
  const float* b_ih0   = (const float*)d_in[3];
  const float* b_hh0   = (const float*)d_in[4];
  const float* w_ih1   = (const float*)d_in[5];
  const float* w_hh1   = (const float*)d_in[6];
  const float* b_ih1   = (const float*)d_in[7];
  const float* b_hh1   = (const float*)d_in[8];
  const float* attn_w1 = (const float*)d_in[9];
  const float* attn_b1 = (const float*)d_in[10];
  const float* attn_w2 = (const float*)d_in[11];
  const float* attn_b2 = (const float*)d_in[12];
  const float* ln_g    = (const float*)d_in[13];
  const float* ln_b    = (const float*)d_in[14];
  const float* head_w1 = (const float*)d_in[15];
  const float* head_b1 = (const float*)d_in[16];
  const float* head_w2 = (const float*)d_in[17];
  const float* head_b2 = (const float*)d_in[18];

  char* ws = (char*)d_ws;
  const size_t HSEQ_BYTES = (size_t)TT * NBT * 2048;   // 64 MiB each
  uint4* h0seq = (uint4*)ws;
  uint4* h1seq = (uint4*)(ws + HSEQ_BYTES);
  float* wlog  = (float*)(ws + 2 * HSEQ_BYTES);
  float* wsm   = (float*)(ws + 2 * HSEQ_BYTES + (size_t)BATCH * TT * 4);

  gru_rec_kernel<0><<<NBT, 128, 0, stream>>>(x, nullptr, w_ih0, w_hh0, b_ih0,
                                             b_hh0, h0seq);
  gru_rec_kernel<1><<<NBT, 128, 0, stream>>>(nullptr, h0seq, w_ih1, w_hh1,
                                             b_ih1, b_hh1, h1seq);
  attn_logits_kernel<<<dim3(NBT, TT / 16), 32, 0, stream>>>(
      h1seq, attn_w1, attn_b1, attn_w2, attn_b2, wlog);
  softmax_kernel<<<BATCH, 256, 0, stream>>>(wlog, wsm);
  pool_head_kernel<<<BATCH, 64, 0, stream>>>(h1seq, wsm, ln_g, ln_b, head_w1,
                                             head_b1, head_w2, head_b2,
                                             (float*)d_out);
}
